// MultiHeadAttention_50929722196636
// MI455X (gfx1250) — matrix operand order repaired and checked by reading, not compiled, not run on hardware
//
#include <hip/hip_runtime.h>
#include <hip/hip_bf16.h>
#include <stdint.h>

#define D_MODEL 1024
#define NUM_HEADS 16
#define SEQ 2048
#define BATCH 4
#define DH 64

typedef __attribute__((ext_vector_type(16))) __bf16 v16bf;
typedef __attribute__((ext_vector_type(8)))  float  v8f;
typedef __attribute__((ext_vector_type(4)))  unsigned int uint4v;
typedef __attribute__((ext_vector_type(2)))  float  f32x2;
typedef __attribute__((ext_vector_type(2)))  __bf16 bf16x2;

#if defined(__has_builtin)
#  if __has_builtin(__builtin_amdgcn_global_load_async_to_lds_b128)
#    define HAS_ASYNC 1
#  endif
#endif
#ifndef HAS_ASYNC
#  define HAS_ASYNC 0
#endif

// Exact pointee type the async-LDS builtins expect (from hipcc diagnostic):
// GNU vector of 4 ints; AS1 (printed as __device__) source, AS3 LDS dest.
typedef int v4i_gnu __attribute__((vector_size(16)));
typedef __attribute__((address_space(1))) v4i_gnu g_v4i;
typedef __attribute__((address_space(3))) v4i_gnu s_v4i;

union BF16Frag {
    v16bf        v;
    __bf16       h[16];
    unsigned int u[8];
    uint4v       q[2];
};

// K index inside a 16x32 bf16 A-fragment for element e (0..15) and lane half.
// ISA 7.12.2: lanes0-15 VGPR0..3 K=0..7, VGPR4..7 K=16..23; lanes16-31 +8.
__device__ __forceinline__ int a_k(int e, int half) {
    int vv = e >> 1, j = e & 1;
    return 2 * vv + ((vv >= 4) ? 8 : 0) + j + half * 8;
}

// packed f32x2 -> bf16x2 (v_cvt_pk_bf16_f32) returned as one dword
__device__ __forceinline__ unsigned cvt_pk2(float x, float y) {
    union { bf16x2 v; unsigned u; } t;
    f32x2 f = {x, y};
    t.v = __builtin_convertvector(f, bf16x2);
    return t.u;
}

// Copy 16 elements of T from global to LDS.
// Async (ASYNCcnt) path when available, else vectorized sync copy.
template <typename T>
__device__ __forceinline__ void copy16_to_lds(const T* __restrict__ g, T* l) {
#if HAS_ASYNC
    constexpr int EPB = 16 / (int)sizeof(T);  // elements per b128
#pragma unroll
    for (int o = 0; o < 16; o += EPB)
        __builtin_amdgcn_global_load_async_to_lds_b128(
            (g_v4i*)(g + o), (s_v4i*)(l + o), 0, 0);
#else
    constexpr int NB = (int)sizeof(T);        // #16B chunks in 16 elements
    const uint4v* s = reinterpret_cast<const uint4v*>(g);
    uint4v*       d = reinterpret_cast<uint4v*>(l);
#pragma unroll
    for (int i = 0; i < NB; ++i) d[i] = s[i];
#endif
}

template <int N>
__device__ __forceinline__ void wait_async_tiles() {
#if HAS_ASYNC
#  if __has_builtin(__builtin_amdgcn_s_wait_asynccnt)
    __builtin_amdgcn_s_wait_asynccnt(N);
#  else
    asm volatile("s_wait_asynccnt %0" ::"i"(N) : "memory");
#  endif
#endif
}

__device__ __forceinline__ void store_one(float* p, float v)  { *p = v; }
__device__ __forceinline__ void store_one(__bf16* p, float v) { *p = (__bf16)v; }

// ===========================================================================
// Tiled WMMA GEMM with double-buffered async tile staging.
// C[:, cofs:cofs+N] = A[MxK] * B[KxN] + bias
// Block = 256 threads (8 waves), tile 128x128, K-step 32 (one WMMA K).
// Waves 2x4; each wave owns 64x32 (4x2 C-frags) -> 8 WMMAs per K-step.
// ===========================================================================
template <typename AT, typename OT>
__global__ __launch_bounds__(256)
void gemm_bias_wmma(const AT* __restrict__ A, const float* __restrict__ B,
                    const float* __restrict__ bias, OT* __restrict__ C,
                    int M, int N, int K, int ldc, int cofs) {
    constexpr int APAD = (sizeof(AT) == 4) ? 4 : 8;      // keep 16B-aligned rows
    __shared__ AT    Ast[2][128][32 + APAD];
    __shared__ float Bst[2][32][128 + 4];

    const int tid  = threadIdx.x;
    const int lane = tid & 31;
    const int wid  = tid >> 5;
    const int half = lane >> 4;
    const int ln   = lane & 15;
    const int wy   = wid >> 2;   // 0..1 -> 64-row slab
    const int wx   = wid & 3;    // 0..3 -> 32-col slab

    const int rblk = blockIdx.y * 128;
    const int cblk = blockIdx.x * 128;

    const v8f zero = {0, 0, 0, 0, 0, 0, 0, 0};
    v8f acc[4][2];
#pragma unroll
    for (int rf = 0; rf < 4; ++rf)
#pragma unroll
        for (int cf = 0; cf < 2; ++cf) acc[rf][cf] = zero;

    const int arow = tid >> 1;          // 0..127
    const int acol = (tid & 1) * 16;    // 0 / 16
    const int brow = tid >> 3;          // 0..31
    const int bcol = (tid & 7) * 16;    // 0..112

    // async ops issued per tile per thread (for s_wait_asynccnt threshold)
    constexpr int NOPS = (sizeof(AT) == 4) ? 8 : 6;

    auto issue_tile = [&](int kt, int buf) {
        const int k0 = kt * 32;
        copy16_to_lds(A + (size_t)(rblk + arow) * K + k0 + acol, &Ast[buf][arow][acol]);
        copy16_to_lds(B + (size_t)(k0 + brow) * N + cblk + bcol, &Bst[buf][brow][bcol]);
    };

    const int KT = K / 32;
    issue_tile(0, 0);

    for (int kt = 0; kt < KT; ++kt) {
        const int cur = kt & 1;
        if (kt + 1 < KT) {
            issue_tile(kt + 1, cur ^ 1);
            wait_async_tiles<NOPS>();   // older tile (ours) has landed
        } else {
            wait_async_tiles<0>();
        }
        __syncthreads();

        BF16Frag af[4];
#pragma unroll
        for (int rf = 0; rf < 4; ++rf) {
            int row = wy * 64 + rf * 16 + ln;
#pragma unroll
            for (int p = 0; p < 8; ++p) {
                int kk = a_k(2 * p, half);  // even -> contiguous K pair
                if constexpr (sizeof(AT) == 4) {
                    f32x2 two = *reinterpret_cast<const f32x2*>(&Ast[cur][row][kk]);
                    af[rf].u[p] = cvt_pk2(two.x, two.y);
                } else {
                    af[rf].u[p] = *reinterpret_cast<const unsigned int*>(&Ast[cur][row][kk]);
                }
            }
        }
        BF16Frag bfr[2];
#pragma unroll
        for (int cf = 0; cf < 2; ++cf) {
            int col = wx * 32 + cf * 16 + ln;
#pragma unroll
            for (int p = 0; p < 8; ++p)
                bfr[cf].u[p] = cvt_pk2(Bst[cur][a_k(2 * p, half)][col],
                                       Bst[cur][a_k(2 * p, half) + 1][col]);
        }
#pragma unroll
        for (int rf = 0; rf < 4; ++rf)
#pragma unroll
            for (int cf = 0; cf < 2; ++cf)
                acc[rf][cf] = __builtin_amdgcn_wmma_f32_16x16x32_bf16(
                    false, af[rf].v, false, bfr[cf].v, (short)0, acc[rf][cf],
                    false, false);
        __syncthreads();
    }

    // epilogue: bias + store
#pragma unroll
    for (int rf = 0; rf < 4; ++rf) {
#pragma unroll
        for (int cf = 0; cf < 2; ++cf) {
            int nloc = cblk + wx * 32 + cf * 16 + ln;
            float bv = bias[nloc];
#pragma unroll
            for (int i = 0; i < 8; ++i) {
                int row = rblk + wy * 64 + rf * 16 + i + half * 8;
                store_one(&C[(size_t)row * ldc + cofs + nloc], acc[rf][cf][i] + bv);
            }
        }
    }
}

// ===========================================================================
// Flash attention: one block = (b, h, 64 queries); 4 waves of 16 queries.
// Keys processed 32 at a time; K/V chunk staged in LDS via async copy.
// ===========================================================================
__global__ __launch_bounds__(128)
void flash_attn_kernel(const __bf16* __restrict__ qkv, __bf16* __restrict__ vals,
                       const int* __restrict__ dmask) {
    const int S    = SEQ;
    const int b    = blockIdx.z;
    const int h    = blockIdx.y;
    const int qblk = blockIdx.x * 64;
    const int tid  = threadIdx.x;
    const int lane = tid & 31;
    const int wave = tid >> 5;
    const int half = lane >> 4;
    const int ln   = lane & 15;

    const bool causal = (dmask[0] != 0);

    __shared__ __bf16 Ks[32][80];       // 160B stride, 16B aligned
    __shared__ __bf16 Vs[32][80];
    __shared__ __bf16 Pb[4][16][32];    // per-wave P bounce

    const int q0   = qblk + wave * 16;
    const int qcol = h * 192;
    const int kcol = h * 192 + 64;      // v is at kcol + 64

    // Q fragments: two 16x32 bf16 A-frags covering dh=64
    BF16Frag qa[2];
    {
        const __bf16* qrow = qkv + (size_t)(b * S + q0 + ln) * 3072 + qcol;
#pragma unroll
        for (int ks = 0; ks < 2; ++ks)
#pragma unroll
            for (int p = 0; p < 8; ++p) {
                int kk = 2 * p + half * 16 + ks * 32;
                qa[ks].u[p] = *reinterpret_cast<const unsigned int*>(qrow + kk);
            }
    }

    float m_i[8], l_i[8];
#pragma unroll
    for (int i = 0; i < 8; ++i) { m_i[i] = -1e30f; l_i[i] = 0.f; }
    const v8f zero = {0, 0, 0, 0, 0, 0, 0, 0};
    v8f oacc[4];
#pragma unroll
    for (int g = 0; g < 4; ++g) oacc[g] = zero;

    const int nchunks = causal ? (qblk / 32 + 2) : (S / 32);

    for (int c = 0; c < nchunks; ++c) {
        const int kbase = c * 32;
        {   // K/V chunk (32 keys x 64) -> LDS, async when available
            int key = tid >> 2;
            int seg = (tid & 3) * 16;
            const __bf16* krow =
                qkv + (size_t)(b * S + kbase + key) * 3072 + kcol + seg;
            copy16_to_lds(krow,      &Ks[key][seg]);
            copy16_to_lds(krow + 64, &Vs[key][seg]);
            wait_async_tiles<0>();
        }
        __syncthreads();

        // scores S16x32 = Q(16x64) * K^T(64x32), two 16-key groups
        v8f sacc[2] = {zero, zero};
#pragma unroll
        for (int ks = 0; ks < 2; ++ks) {
#pragma unroll
            for (int grp = 0; grp < 2; ++grp) {
                BF16Frag bk;
                int key = grp * 16 + ln;
                const uint4v* kp =
                    reinterpret_cast<const uint4v*>(&Ks[key][ks * 32 + half * 16]);
                bk.q[0] = kp[0];
                bk.q[1] = kp[1];
                sacc[grp] = __builtin_amdgcn_wmma_f32_16x16x32_bf16(
                    false, qa[ks].v, false, bk.v, (short)0, sacc[grp], false, false);
            }
        }

        const float scale = 0.125f;  // 1/sqrt(64)
        float s0[8], s1[8];
#pragma unroll
        for (int i = 0; i < 8; ++i) {
            int row = q0 + i + half * 8;
            float a  = sacc[0][i] * scale;
            float bb = sacc[1][i] * scale;
            if (causal) {
                if (kbase + ln > row)      a  = -1e30f;
                if (kbase + 16 + ln > row) bb = -1e30f;
            }
            s0[i] = a; s1[i] = bb;
        }
#pragma unroll
        for (int i = 0; i < 8; ++i) {
            float cm = fmaxf(s0[i], s1[i]);
            cm = fmaxf(cm, __shfl_xor(cm, 1, 32));
            cm = fmaxf(cm, __shfl_xor(cm, 2, 32));
            cm = fmaxf(cm, __shfl_xor(cm, 4, 32));
            cm = fmaxf(cm, __shfl_xor(cm, 8, 32));
            float mn    = fmaxf(m_i[i], cm);
            float alpha = __expf(m_i[i] - mn);
            float p0 = __expf(s0[i] - mn);
            float p1 = __expf(s1[i] - mn);
            float rs = p0 + p1;
            rs += __shfl_xor(rs, 1, 32);
            rs += __shfl_xor(rs, 2, 32);
            rs += __shfl_xor(rs, 4, 32);
            rs += __shfl_xor(rs, 8, 32);
            l_i[i] = l_i[i] * alpha + rs;
            m_i[i] = mn;
#pragma unroll
            for (int g = 0; g < 4; ++g) oacc[g][i] *= alpha;
            s0[i] = p0; s1[i] = p1;
        }
        // P (C-layout) -> per-wave LDS -> A-layout (same-wave DS is in-order)
#pragma unroll
        for (int i = 0; i < 8; ++i) {
            int r = i + half * 8;
            Pb[wave][r][ln]      = (__bf16)s0[i];
            Pb[wave][r][16 + ln] = (__bf16)s1[i];
        }
        asm volatile("s_wait_dscnt 0" ::: "memory");
        BF16Frag pa;
#pragma unroll
        for (int p = 0; p < 8; ++p) {
            int kk = a_k(2 * p, half);
            pa.u[p] = *reinterpret_cast<const unsigned int*>(&Pb[wave][ln][kk]);
        }
        // O(16x64) += P(16x32) * V(32x64)
#pragma unroll
        for (int g = 0; g < 4; ++g) {
            BF16Frag bv;
            int col = g * 16 + ln;
#pragma unroll
            for (int e = 0; e < 16; ++e) bv.h[e] = Vs[a_k(e, half)][col];
            oacc[g] = __builtin_amdgcn_wmma_f32_16x16x32_bf16(
                false, pa.v, false, bv.v, (short)0, oacc[g], false, false);
        }
        __syncthreads();
    }

    // normalize + store vals (bf16)
#pragma unroll
    for (int i = 0; i < 8; ++i) {
        int row = q0 + i + half * 8;
        float inv = 1.0f / l_i[i];
#pragma unroll
        for (int g = 0; g < 4; ++g) {
            vals[(size_t)(b * S + row) * D_MODEL + h * DH + g * 16 + ln] =
                (__bf16)(oacc[g][i] * inv);
        }
    }
}

// ===========================================================================
extern "C" void kernel_launch(void* const* d_in, const int* in_sizes, int n_in,
                              void* d_out, int out_size, void* d_ws, size_t ws_size,
                              hipStream_t stream) {
    const float* x   = (const float*)d_in[0];
    const float* Wq  = (const float*)d_in[1];
    const float* bq  = (const float*)d_in[2];
    const float* Wkv = (const float*)d_in[3];
    const float* bkv = (const float*)d_in[4];
    const float* Wo  = (const float*)d_in[5];
    const float* bo  = (const float*)d_in[6];
    const int*   dm  = (const int*)d_in[7];
    float* out = (float*)d_out;

    const int M = BATCH * SEQ;  // 8192
    __bf16* qkv   = (__bf16*)d_ws;                                  // 48 MB
    __bf16* valsb = (__bf16*)((char*)d_ws + (size_t)M * 3072 * 2);  // 16 MB

    // QKV projection (two column ranges of one 3072-wide buffer)
    gemm_bias_wmma<float, __bf16><<<dim3(1024 / 128, M / 128), dim3(256), 0, stream>>>(
        x, Wq, bq, qkv, M, 1024, 1024, 3072, 0);
    gemm_bias_wmma<float, __bf16><<<dim3(2048 / 128, M / 128), dim3(256), 0, stream>>>(
        x, Wkv, bkv, qkv, M, 2048, 1024, 3072, 1024);

    // causal flash attention
    flash_attn_kernel<<<dim3(SEQ / 64, NUM_HEADS, BATCH), dim3(128), 0, stream>>>(
        qkv, valsb, dm);

    // output projection
    gemm_bias_wmma<__bf16, float><<<dim3(1024 / 128, M / 128), dim3(256), 0, stream>>>(
        valsb, Wo, bo, out, M, 1024, 1024, 1024, 0);
}
